// BiLSTM_CRF_61976378081807
// MI455X (gfx1250) — matrix operand order repaired and checked by reading, not compiled, not run on hardware
//
#include <hip/hip_runtime.h>
#include <hip/hip_bf16.h>
#include <math.h>

typedef __attribute__((ext_vector_type(16))) _Float16 v16h;
typedef __attribute__((ext_vector_type(8)))  float    v8f;

#define B_    64
#define L_    256
#define E_    256
#define HH_   256
#define NG_   1024   // 4*HH
#define KT_X  8      // K-subtiles covering x (E=256 / 32)
#define KT_ALL 16    // K-subtiles covering [x,h] (512 / 32)
#define NT_   64     // N tiles (1024/16)
#define T_    12
#define START_ 10
#define STOP_  11

// ---------------------------------------------------------------------------
// Pack combined weights [W_ih | W_hh] (N=1024 rows, K=512 cols) into f16
// B-fragment tiles: Bpack[dir][nt][kt][512], element e = lane*16 + j with
// n = nt*16 + lane%16, half = lane/16,
// k = kt*32 + (j<8 ? half*8+j : 16+half*8+(j-8)). Also combined bias.
// ---------------------------------------------------------------------------
__global__ void pack_w_kernel(const float* __restrict__ wih0, const float* __restrict__ whh0,
                              const float* __restrict__ bih0, const float* __restrict__ bhh0,
                              const float* __restrict__ wih1, const float* __restrict__ whh1,
                              const float* __restrict__ bih1, const float* __restrict__ bhh1,
                              _Float16* __restrict__ Bpack, float* __restrict__ biasc)
{
    int id = blockIdx.x * blockDim.x + threadIdx.x;   // 2^20 total
    int e   = id & 511;
    int kt  = (id >> 9)  & 15;
    int nt  = (id >> 13) & 63;
    int dir = (id >> 19) & 1;
    int lane = e >> 4, j = e & 15;
    int n = nt * 16 + (lane & 15);
    int k = kt * 32 + ((j < 8) ? ((lane >> 4) * 8 + j) : (16 + (lane >> 4) * 8 + (j - 8)));
    const float* wih = dir ? wih1 : wih0;
    const float* whh = dir ? whh1 : whh0;
    float v = (k < E_) ? wih[n * E_ + k] : whh[n * HH_ + (k - E_)];
    Bpack[id] = (_Float16)v;
    if (id < 2 * NG_) {
        int d2 = id >> 10, n2 = id & (NG_ - 1);
        const float* bi = d2 ? bih1 : bih0;
        const float* bh = d2 ? bhh1 : bhh0;
        biasc[id] = bi[n2] + bh[n2];
    }
}

// ---------------------------------------------------------------------------
// Gather embeddings and pre-pack as A-fragments (16x32 f16 per ISA layout):
// Apack[chunk][t][kt][512], e = lane*16+j; m = lane%16, half = lane/16,
// k32 = j<8 ? half*8+j : 16 + half*8 + (j-8).
// ---------------------------------------------------------------------------
__global__ void pack_emb_kernel(const int* __restrict__ sent,
                                const float* __restrict__ emb,
                                _Float16* __restrict__ Apack)
{
    int id = blockIdx.x * blockDim.x + threadIdx.x;   // 4*256*8*512 = 2^22 total
    int e     = id & 511;
    int kt    = (id >> 9)  & 7;
    int t     = (id >> 12) & 255;
    int chunk = (id >> 20) & 3;
    int lane = e >> 4, j = e & 15;
    int m = lane & 15, half = lane >> 4;
    int k32 = (j < 8) ? (half * 8 + j) : (16 + half * 8 + (j - 8));
    int k = kt * 32 + k32;
    int b = chunk * 16 + m;
    int tok = sent[b * L_ + t];
    Apack[id] = (_Float16)emb[(long long)tok * E_ + k];
}

// h -> f16 A-fragment scatter helper (inverse of the pack_emb mapping)
__device__ __forceinline__ void pack_h_lds(_Float16* hA, int row, int hh, float v)
{
    int kt = hh >> 5, k32 = hh & 31;
    int half = (k32 >> 3) & 1;
    int ln = row + (half << 4);
    int jj = (k32 < 16) ? (k32 & 7) : (8 + (k32 & 7));
    hA[kt * 512 + ln * 16 + jj] = (_Float16)v;
}

// ---------------------------------------------------------------------------
// Persistent bidirectional LSTM scan. grid = 8 (dir x 4 batch-chunks of 16
// rows), block = 1024 (32 waves). Each wave owns 2 N-tiles which SHARE one
// A fragment per K-step: load a, b0, b1 -> two WMMAs into two accumulators.
// Weight loads use a base offset laundered through an opaque asm each step,
// so they stay un-hoisted (no 512-VGPR live range / scratch spill) but keep
// their inferred GLOBAL address space -> plain cached global_load_b128.
// ---------------------------------------------------------------------------
__global__ __launch_bounds__(1024) void scan_kernel(
    const _Float16* __restrict__ Apack, const _Float16* __restrict__ Bpack,
    const float* __restrict__ biasc, const float* __restrict__ h0,
    const float* __restrict__ c0, float* __restrict__ hsumG)
{
    __shared__ float    gates[16 * NG_];                             // 64 KB
    __shared__ __attribute__((aligned(32))) _Float16 hA[8 * 512];    //  8 KB

    int dir = blockIdx.x & 1, chunk = blockIdx.x >> 1;
    int tid = threadIdx.x, wave = tid >> 5, lane = tid & 31;
    const float* bC = biasc + dir * NG_;

    float cs[4], hs[4], bi[4], bf[4], bg[4], bo[4];
    int rowA[4], hhA[4];
    #pragma unroll
    for (int i = 0; i < 4; ++i) {
        int idx = tid + i * 1024;
        int row = idx >> 8, hh = idx & 255;
        rowA[i] = row; hhA[i] = hh;
        int b = chunk * 16 + row;
        float hv = h0[(b * 2 + dir) * HH_ + hh];
        cs[i] = c0[(b * 2 + dir) * HH_ + hh];
        hs[i] = 0.0f;
        bi[i] = bC[hh]; bf[i] = bC[256 + hh]; bg[i] = bC[512 + hh]; bo[i] = bC[768 + hh];
        pack_h_lds(hA, row, hh, hv);
    }
    __syncthreads();

    const _Float16* Bd = Bpack + (size_t)dir * NT_ * KT_ALL * 512;
    int nt0 = wave * 2;

    for (int s = 0; s < L_; ++s) {
        int t = dir ? (L_ - 1 - s) : s;
        const _Float16* At = Apack + ((size_t)(chunk * L_ + t) * KT_X) * 512;

        // Opaque zero offset, re-defined every step: blocks LICM/CSE of the
        // weight loads while preserving global address-space inference.
        size_t z = 0;
        asm volatile("" : "+s"(z));
        const _Float16* B0 = Bd + z + (size_t)(nt0 + 0) * KT_ALL * 512;
        const _Float16* B1 = Bd + z + (size_t)(nt0 + 1) * KT_ALL * 512;

        // Prefetch next step's pre-packed x tiles (8 KB) into cache while
        // this step's WMMA work runs.
        if (s + 1 < L_) {
            int t2 = dir ? (L_ - 2 - s) : (s + 1);
            const char* Anext =
                (const char*)(Apack + ((size_t)(chunk * L_ + t2) * KT_X) * 512);
            if (tid < 64) __builtin_prefetch(Anext + tid * 128, 0, 3);
        }

        v8f acc0 = {0.f, 0.f, 0.f, 0.f, 0.f, 0.f, 0.f, 0.f};
        v8f acc1 = {0.f, 0.f, 0.f, 0.f, 0.f, 0.f, 0.f, 0.f};
        #pragma unroll
        for (int kt = 0; kt < KT_ALL; ++kt) {
            v16h a = (kt < KT_X)
                   ? *(const v16h*)(At + kt * 512 + lane * 16)
                   : *(const v16h*)(&hA[(kt - KT_X) * 512 + lane * 16]);
            v16h b0 = *(const v16h*)(B0 + kt * 512 + lane * 16);
            v16h b1 = *(const v16h*)(B1 + kt * 512 + lane * 16);
            acc0 = __builtin_amdgcn_wmma_f32_16x16x32_f16(
                       false, a, false, b0, (short)0, acc0, false, false);
            acc1 = __builtin_amdgcn_wmma_f32_16x16x32_f16(
                       false, a, false, b1, (short)0, acc1, false, false);
        }
        {
            int colb = (lane & 15);
            int rb   = (lane >> 4) * 8;
            #pragma unroll
            for (int r = 0; r < 8; ++r) {
                gates[(rb + r) * NG_ + (nt0 + 0) * 16 + colb] = acc0[r];
                gates[(rb + r) * NG_ + (nt0 + 1) * 16 + colb] = acc1[r];
            }
        }
        __syncthreads();

        #pragma unroll
        for (int i = 0; i < 4; ++i) {
            int row = rowA[i], hh = hhA[i];
            const float* g = gates + row * NG_;
            float gi = 1.0f / (1.0f + __expf(-(g[hh]        + bi[i])));
            float gf = 1.0f / (1.0f + __expf(-(g[256 + hh]  + bf[i])));
            float gg = tanhf(           g[512 + hh]  + bg[i]);
            float go = 1.0f / (1.0f + __expf(-(g[768 + hh]  + bo[i])));
            float c2 = gf * cs[i] + gi * gg;
            float h2 = go * tanhf(c2);
            cs[i] = c2;
            hs[i] += h2;
            pack_h_lds(hA, row, hh, h2);
        }
        __syncthreads();
    }

    #pragma unroll
    for (int i = 0; i < 4; ++i) {
        int b = chunk * 16 + rowA[i];
        hsumG[b * (2 * HH_) + dir * HH_ + hhA[i]] = hs[i] * (1.0f / (float)L_);
    }
}

// ---------------------------------------------------------------------------
// feats[b][tag] = mean-pooled hidden . w_tag[tag] + b_tag[tag]   (B=64, T=12)
// ---------------------------------------------------------------------------
__global__ void feats_kernel(const float* __restrict__ hsumG,
                             const float* __restrict__ wtag,
                             const float* __restrict__ btag,
                             float* __restrict__ feats)
{
    int b = blockIdx.x;
    int tag = threadIdx.x;
    if (tag >= T_) return;
    const float* v = hsumG + b * (2 * HH_);
    const float* w = wtag + tag * (2 * HH_);
    float acc = btag[tag];
    for (int k = 0; k < 2 * HH_; ++k) acc += v[k] * w[k];
    feats[b * T_ + tag] = acc;
}

// ---------------------------------------------------------------------------
// Viterbi over the 64 feature rows, T=12 tags. One wave; lanes = next-tags.
// Output: d_out[0] = path_score, d_out[1..64] = tag_seq (as floats).
// ---------------------------------------------------------------------------
__global__ void viterbi_kernel(const float* __restrict__ feats,
                               const float* __restrict__ trans,
                               float* __restrict__ out)
{
    __shared__ float fv[T_];
    __shared__ int   bps[B_][T_];
    int tid = threadIdx.x;
    if (tid < T_) fv[tid] = (tid == START_) ? 0.0f : -10000.0f;
    __syncthreads();

    for (int step = 0; step < B_; ++step) {
        float best = -3.4e38f; int bp = 0;
        if (tid < T_) {
            for (int p = 0; p < T_; ++p) {
                float sc = fv[p] + trans[tid * T_ + p];
                if (sc > best) { best = sc; bp = p; }
            }
            bps[step][tid] = bp;
        }
        __syncthreads();
        if (tid < T_) fv[tid] = best + feats[step * T_ + tid];
        __syncthreads();
    }

    if (tid == 0) {
        float bestv = -3.4e38f; int bt = 0;
        for (int tg = 0; tg < T_; ++tg) {
            float tv = fv[tg] + trans[STOP_ * T_ + tg];
            if (tv > bestv) { bestv = tv; bt = tg; }
        }
        out[0] = bestv;
        int q[B_];
        int cur = bt;
        for (int s2 = 0; s2 < B_; ++s2) { cur = bps[B_ - 1 - s2][cur]; q[s2] = cur; }
        out[1 + (B_ - 1)] = (float)bt;
        for (int i2 = 0; i2 < B_ - 1; ++i2) out[1 + i2] = (float)q[B_ - 2 - i2];
    }
}

// ---------------------------------------------------------------------------
extern "C" void kernel_launch(void* const* d_in, const int* in_sizes, int n_in,
                              void* d_out, int out_size, void* d_ws, size_t ws_size,
                              hipStream_t stream)
{
    const int*   sent  = (const int*)  d_in[0];
    // d_in[1] = sentence_lens (unused; reference ignores it)
    const float* emb   = (const float*)d_in[2];
    const float* wih0  = (const float*)d_in[3];
    const float* whh0  = (const float*)d_in[4];
    const float* bih0  = (const float*)d_in[5];
    const float* bhh0  = (const float*)d_in[6];
    const float* wih1  = (const float*)d_in[7];
    const float* whh1  = (const float*)d_in[8];
    const float* bih1  = (const float*)d_in[9];
    const float* bhh1  = (const float*)d_in[10];
    const float* wtag  = (const float*)d_in[11];
    const float* btag  = (const float*)d_in[12];
    const float* trans = (const float*)d_in[13];
    const float* h0    = (const float*)d_in[14];
    const float* c0    = (const float*)d_in[15];
    float* out = (float*)d_out;

    char* ws = (char*)d_ws;
    _Float16* Bpack = (_Float16*)ws;                                  // 2 MB
    size_t off = 2ull * NT_ * KT_ALL * 512 * sizeof(_Float16);
    float* biasc = (float*)(ws + off);  off += 2ull * NG_ * sizeof(float);
    _Float16* Apack = (_Float16*)(ws + off);                          // 8 MB
    off += 4ull * L_ * KT_X * 512 * sizeof(_Float16);
    float* hsumG = (float*)(ws + off);  off += (size_t)B_ * 2 * HH_ * sizeof(float);
    float* feats = (float*)(ws + off);  off += (size_t)B_ * T_ * sizeof(float);

    pack_w_kernel  <<<4096,  256, 0, stream>>>(wih0, whh0, bih0, bhh0,
                                               wih1, whh1, bih1, bhh1, Bpack, biasc);
    pack_emb_kernel<<<16384, 256, 0, stream>>>(sent, emb, Apack);
    scan_kernel    <<<8,    1024, 0, stream>>>(Apack, Bpack, biasc, h0, c0, hsumG);
    feats_kernel   <<<B_,     16, 0, stream>>>(hsumG, wtag, btag, feats);
    viterbi_kernel <<<1,      32, 0, stream>>>(feats, trans, out);
}
